// QwenDoubleStreamAttentionMergedQKV_40149354283479
// MI455X (gfx1250) — compile-verified, matrix operand order checked by reading, not run on hardware
//
#include <hip/hip_runtime.h>

#define DIMC   3072
#define HEADS  24
#define HD     128
#define S_IMG  2048
#define S_TXT  256
#define S_TOT  2304
#define TDIM   9216   // 3*DIM

typedef __bf16 bf16_t;
typedef __attribute__((ext_vector_type(16))) __bf16 v16bf;
typedef __attribute__((ext_vector_type(8)))  __bf16 bf16x8;
typedef __attribute__((ext_vector_type(8)))  float  v8f;

__device__ __forceinline__ bf16_t f2bf(float f) {
  union { float f; unsigned u; } v; v.f = f;
  unsigned r = v.u + 0x7FFFu + ((v.u >> 16) & 1u);   // round-to-nearest-even
  union { unsigned short s; bf16_t b; } o; o.s = (unsigned short)(r >> 16);
  return o.b;
}

// Async 16-byte global -> LDS copy (ASYNCcnt-tracked, bypasses VGPRs).
// Generic LDS pointers carry the LDS byte offset in their low 32 bits.
__device__ __forceinline__ void async_b128(void* lds, const void* g) {
  unsigned l = (unsigned)(uintptr_t)lds;
  asm volatile("global_load_async_to_lds_b128 %0, %1, off"
               :: "v"(l), "v"(g) : "memory");
}
__device__ __forceinline__ void wait_async0() {
  asm volatile("s_wait_asynccnt 0x0" ::: "memory");
}

// ---------------------------------------------------------------- convert
__global__ void k_f32_to_bf16(const float* __restrict__ in,
                              bf16_t* __restrict__ out, int n) {
  int i = blockIdx.x * blockDim.x + threadIdx.x;
  int stride = gridDim.x * blockDim.x;
  for (; i < n; i += stride) out[i] = f2bf(in[i]);
}

// ------------------------------------------------- convert + transpose
// in[R][C] fp32 -> out[C][R] bf16, 32x32 LDS tiles, 256 threads (32x8).
__global__ __launch_bounds__(256) void k_f32_to_bf16_T(
    const float* __restrict__ in, bf16_t* __restrict__ out, int R, int C) {
  __shared__ float t[32][33];
  const int bx = blockIdx.x * 32;             // col tile (C)
  const int by = blockIdx.y * 32;             // row tile (R)
  const int tx = threadIdx.x & 31, ty = threadIdx.x >> 5;
#pragma unroll
  for (int i = 0; i < 32; i += 8)
    t[ty + i][tx] = in[(size_t)(by + ty + i) * C + bx + tx];
  __syncthreads();
#pragma unroll
  for (int i = 0; i < 32; i += 8)
    out[(size_t)(bx + ty + i) * R + by + tx] = f2bf(t[tx][ty + i]);
}

// ---------------------------------------------------------------- GEMM
// C[M,N] = A[M,K](bf16,row) * Bt[N,K](bf16, pre-transposed) + bias[N].
// 256 threads = 8 waves; tile 128x128, K-step 32; double-buffered LDS with
// async global->LDS staging overlapped against ds_load + WMMA compute.
__global__ __launch_bounds__(256) void k_gemm_bf16(
    const bf16_t* __restrict__ A, const bf16_t* __restrict__ Bt,
    const float* __restrict__ bias, float* __restrict__ C,
    int M, int N, int K) {
  __shared__ __attribute__((aligned(32))) bf16_t As[2][128 * 32];
  __shared__ __attribute__((aligned(32))) bf16_t Bs[2][128 * 32];
  const int tid  = threadIdx.x;
  const int lane = tid & 31;
  const int w    = tid >> 5;
  const int wm   = w >> 1, wn = w & 1;
  const int m0   = blockIdx.y * 128;
  const int n0   = blockIdx.x * 128;

  auto stage = [&](int k0, int buf) {
#pragma unroll
    for (int i = 0; i < 2; ++i) {
      const int c = tid * 2 + i;               // 0..511
      const int row = c >> 2, k8 = c & 3;      // 128 rows x 4 chunks of 8
      async_b128(&As[buf][row * 32 + k8 * 8],
                 A + (size_t)(m0 + row) * K + k0 + k8 * 8);
      async_b128(&Bs[buf][row * 32 + k8 * 8],
                 Bt + (size_t)(n0 + row) * K + k0 + k8 * 8);
    }
  };

  v8f acc[2][4];
  const v8f vzero = {0.f,0.f,0.f,0.f,0.f,0.f,0.f,0.f};
#pragma unroll
  for (int i = 0; i < 2; ++i)
#pragma unroll
    for (int j = 0; j < 4; ++j) acc[i][j] = vzero;

  stage(0, 0);
  wait_async0();
  __syncthreads();

  for (int k0 = 0; k0 < K; k0 += 32) {
    const int cur = (k0 >> 5) & 1;
    if (k0 + 32 < K) stage(k0 + 32, cur ^ 1);   // prefetch next tile (async)

    v16bf af[2];
#pragma unroll
    for (int mi = 0; mi < 2; ++mi)
      af[mi] = *(const v16bf*)(&As[cur][(wm * 32 + mi * 16 + (lane & 15)) * 32 +
                                        (lane >> 4) * 16]);
#pragma unroll
    for (int ni = 0; ni < 4; ++ni) {
      v16bf bfr = *(const v16bf*)(&Bs[cur][(wn * 64 + ni * 16 + (lane & 15)) * 32 +
                                           (lane >> 4) * 16]);
#pragma unroll
      for (int mi = 0; mi < 2; ++mi)
        acc[mi][ni] = __builtin_amdgcn_wmma_f32_16x16x32_bf16(
            false, af[mi], false, bfr, (short)0, acc[mi][ni], false, false);
    }

    wait_async0();
    __syncthreads();
  }

  // Epilogue: D layout lane L -> col = L&15, VGPR r -> row = r + 8*(L>>4)
#pragma unroll
  for (int mi = 0; mi < 2; ++mi)
#pragma unroll
    for (int ni = 0; ni < 4; ++ni) {
      const int col = n0 + wn * 64 + ni * 16 + (lane & 15);
      const float bc = bias[col];
#pragma unroll
      for (int r = 0; r < 8; ++r) {
        const int row = m0 + wm * 32 + mi * 16 + r + ((lane >> 4) << 3);
        C[(size_t)row * N + col] = acc[mi][ni][r] + bc;
      }
    }
}

// ---------------------------------------------------------------- prep: RMS norm + RoPE
// grid (S_TOT, HEADS), 128 threads. q/k -> [head][seq][128]; v -> [head][128][seq]
// (transposed so the attention P*V B-operand stages contiguously).
__global__ __launch_bounds__(128) void k_prep_qkv(
    const float* __restrict__ img_qkv, const float* __restrict__ txt_qkv,
    const float* __restrict__ img_cos, const float* __restrict__ img_sin,
    const float* __restrict__ txt_cos, const float* __restrict__ txt_sin,
    const float* __restrict__ gq,  const float* __restrict__ gk,
    const float* __restrict__ gaq, const float* __restrict__ gak,
    bf16_t* __restrict__ qb, bf16_t* __restrict__ kb, bf16_t* __restrict__ vb) {
  __shared__ float red[128];
  __shared__ float sq[128];
  __shared__ float sk[128];
  const int s = blockIdx.x, h = blockIdx.y, d = threadIdx.x;
  const bool is_txt = (s < S_TXT);
  const int pos = is_txt ? s : (s - S_TXT);
  const float* base = is_txt ? (txt_qkv + (size_t)pos * TDIM)
                             : (img_qkv + (size_t)pos * TDIM);
  const float* ct = is_txt ? txt_cos : img_cos;
  const float* st = is_txt ? txt_sin : img_sin;

  const float qv = base[h * HD + d];
  const float kv = base[DIMC + h * HD + d];
  const float vv = base[2 * DIMC + h * HD + d];

  red[d] = qv * qv; __syncthreads();
  for (int off = 64; off > 0; off >>= 1) {
    if (d < off) red[d] += red[d + off];
    __syncthreads();
  }
  const float rq = rsqrtf(red[0] * (1.0f / HD) + 1e-6f);
  __syncthreads();
  red[d] = kv * kv; __syncthreads();
  for (int off = 64; off > 0; off >>= 1) {
    if (d < off) red[d] += red[d + off];
    __syncthreads();
  }
  const float rk = rsqrtf(red[0] * (1.0f / HD) + 1e-6f);
  __syncthreads();

  sq[d] = qv * rq * (is_txt ? gaq[d] : gq[d]);
  sk[d] = kv * rk * (is_txt ? gak[d] : gk[d]);
  __syncthreads();

  const int i = d >> 1;
  const float c = ct[pos * (HD / 2) + i], sn = st[pos * (HD / 2) + i];
  float qo, ko;
  if ((d & 1) == 0) { qo = sq[d] * c - sq[d + 1] * sn;
                      ko = sk[d] * c - sk[d + 1] * sn; }
  else              { qo = sq[d - 1] * sn + sq[d] * c;
                      ko = sk[d - 1] * sn + sk[d] * c; }

  const size_t oi = ((size_t)h * S_TOT + s) * HD + d;
  qb[oi] = f2bf(qo);
  kb[oi] = f2bf(ko);
  vb[((size_t)h * HD + d) * S_TOT + s] = f2bf(vv);
}

// ---------------------------------------------------------------- attention
// grid (S_TOT/64, HEADS), 128 threads = 4 waves; each wave 16 q rows x D=128.
// Flash-style online softmax; K/V tiles double-buffered via async LDS copies.
__global__ __launch_bounds__(128) void k_attention(
    const bf16_t* __restrict__ qb, const bf16_t* __restrict__ kb,
    const bf16_t* __restrict__ vb, float* __restrict__ attn) {
  __shared__ __attribute__((aligned(32))) bf16_t Qs[64 * 128];
  __shared__ __attribute__((aligned(32))) bf16_t Ks[2][64 * 128]; // [s_k][d]
  __shared__ __attribute__((aligned(32))) bf16_t Vt[2][128 * 64]; // [d][s_k]
  __shared__ __attribute__((aligned(32))) bf16_t Ps[4][16 * 64];
  const int tid = threadIdx.x, lane = tid & 31, w = tid >> 5;
  const int h  = blockIdx.y;
  const int q0 = blockIdx.x * 64;
  const bf16_t* Qg  = qb + (size_t)h * S_TOT * HD;
  const bf16_t* Kg  = kb + (size_t)h * S_TOT * HD;
  const bf16_t* VgT = vb + (size_t)h * HD * S_TOT;   // [d][s]

  auto stage_kv = [&](int kt, int buf) {
    const int k0 = kt * 64;
#pragma unroll
    for (int i = 0; i < 8; ++i) {
      // K: 64 rows x 16 chunks of 8
      {
        const int c = tid * 8 + i;
        const int row = c >> 4, ch = c & 15;
        async_b128(&Ks[buf][row * 128 + ch * 8],
                   Kg + (size_t)(k0 + row) * HD + ch * 8);
      }
      // V^T: 128 d-rows x 8 chunks of 8
      {
        const int c = tid * 8 + i;
        const int dr = c >> 3, ch = c & 7;
        async_b128(&Vt[buf][dr * 64 + ch * 8],
                   VgT + (size_t)dr * S_TOT + k0 + ch * 8);
      }
    }
  };

  // stage Q (async) + first K/V tile
#pragma unroll
  for (int i = 0; i < 8; ++i) {
    const int c = tid * 8 + i;
    const int row = c >> 4, ch = c & 15;
    async_b128(&Qs[row * 128 + ch * 8], Qg + (size_t)(q0 + row) * HD + ch * 8);
  }
  stage_kv(0, 0);
  wait_async0();
  __syncthreads();

  v16bf qf[4];
#pragma unroll
  for (int kk = 0; kk < 4; ++kk)
    qf[kk] = *(const v16bf*)(Qs + (w * 16 + (lane & 15)) * 128 + kk * 32 +
                             (lane >> 4) * 16);

  const v8f vzero = {0.f,0.f,0.f,0.f,0.f,0.f,0.f,0.f};
  float m_i[8], l_i[8];
  v8f accd[8];
#pragma unroll
  for (int r = 0; r < 8; ++r) { m_i[r] = -1e30f; l_i[r] = 0.f; }
#pragma unroll
  for (int dt = 0; dt < 8; ++dt) accd[dt] = vzero;

  const float scale = 0.08838834764831845f;   // 1/sqrt(128)
  const int NT = S_TOT / 64;

  for (int kt = 0; kt < NT; ++kt) {
    const int cur = kt & 1;
    if (kt + 1 < NT) stage_kv(kt + 1, cur ^ 1);   // async prefetch next tile

    // S = Q * K^T  (16x64 per wave)
    v8f sc[4];
#pragma unroll
    for (int nt = 0; nt < 4; ++nt) {
      sc[nt] = vzero;
#pragma unroll
      for (int kk = 0; kk < 4; ++kk) {
        v16bf bfr = *(const v16bf*)(&Ks[cur][(nt * 16 + (lane & 15)) * 128 +
                                             kk * 32 + (lane >> 4) * 16]);
        sc[nt] = __builtin_amdgcn_wmma_f32_16x16x32_bf16(
            false, qf[kk], false, bfr, (short)0, sc[nt], false, false);
      }
      sc[nt] = sc[nt] * scale;
    }

    // online softmax: row lives in one 16-lane half; reduce with shfl_xor
#pragma unroll
    for (int r = 0; r < 8; ++r) {
      float mloc = fmaxf(fmaxf(sc[0][r], sc[1][r]), fmaxf(sc[2][r], sc[3][r]));
#pragma unroll
      for (int x = 1; x < 16; x <<= 1) mloc = fmaxf(mloc, __shfl_xor(mloc, x, 32));
      const float mnew  = fmaxf(m_i[r], mloc);
      const float alpha = __expf(m_i[r] - mnew);
      m_i[r] = mnew;
      float rs = 0.f;
#pragma unroll
      for (int nt = 0; nt < 4; ++nt) {
        const float p = __expf(sc[nt][r] - mnew);
        sc[nt][r] = p; rs += p;
      }
#pragma unroll
      for (int x = 1; x < 16; x <<= 1) rs += __shfl_xor(rs, x, 32);
      l_i[r] = l_i[r] * alpha + rs;
#pragma unroll
      for (int dt = 0; dt < 8; ++dt) accd[dt][r] *= alpha;
    }

    // D-layout -> A-layout via per-wave LDS tile (wave-local DS is in-order)
#pragma unroll
    for (int nt = 0; nt < 4; ++nt)
#pragma unroll
      for (int r = 0; r < 8; ++r)
        Ps[w][(r + ((lane >> 4) << 3)) * 64 + nt * 16 + (lane & 15)] =
            f2bf(sc[nt][r]);

    // O += P(16x64) * V(64x128)
#pragma unroll
    for (int kh = 0; kh < 2; ++kh) {
      v16bf pf = *(const v16bf*)(&Ps[w][(lane & 15) * 64 + kh * 32 +
                                        (lane >> 4) * 16]);
#pragma unroll
      for (int dt = 0; dt < 8; ++dt) {
        v16bf vfr = *(const v16bf*)(&Vt[cur][(dt * 16 + (lane & 15)) * 64 +
                                             kh * 32 + (lane >> 4) * 16]);
        accd[dt] = __builtin_amdgcn_wmma_f32_16x16x32_bf16(
            false, pf, false, vfr, (short)0, accd[dt], false, false);
      }
    }

    wait_async0();
    __syncthreads();
  }

#pragma unroll
  for (int dt = 0; dt < 8; ++dt)
#pragma unroll
    for (int r = 0; r < 8; ++r) {
      const int row = q0 + w * 16 + r + ((lane >> 4) << 3);
      const int col = h * HD + dt * 16 + (lane & 15);
      attn[(size_t)row * DIMC + col] = accd[dt][r] / l_i[r];
    }
}

// ---------------------------------------------------------------- launch
extern "C" void kernel_launch(void* const* d_in, const int* in_sizes, int n_in,
                              void* d_out, int out_size, void* d_ws, size_t ws_size,
                              hipStream_t stream) {
  (void)in_sizes; (void)n_in; (void)out_size; (void)ws_size;
  const float* image    = (const float*)d_in[0];
  const float* text     = (const float*)d_in[1];
  const float* img_cos  = (const float*)d_in[2];
  const float* img_sin  = (const float*)d_in[3];
  const float* txt_cos  = (const float*)d_in[4];
  const float* txt_sin  = (const float*)d_in[5];
  const float* Wqkv     = (const float*)d_in[6];
  const float* bqkv     = (const float*)d_in[7];
  const float* Wadd     = (const float*)d_in[8];
  const float* badd     = (const float*)d_in[9];
  const float* gq       = (const float*)d_in[10];
  const float* gk       = (const float*)d_in[11];
  const float* gaq      = (const float*)d_in[12];
  const float* gak      = (const float*)d_in[13];
  const float* Wout     = (const float*)d_in[14];
  const float* bout     = (const float*)d_in[15];
  const float* Waddout  = (const float*)d_in[16];
  const float* baddout  = (const float*)d_in[17];
  float* out = (float*)d_out;

  char* ws = (char*)d_ws;
  size_t off = 0;
  auto alloc = [&](size_t bytes) -> void* {
    off = (off + 255) & ~(size_t)255;
    void* p = ws + off;
    off += bytes;
    return p;
  };

  bf16_t* Xib       = (bf16_t*)alloc((size_t)S_IMG * DIMC * 2);
  bf16_t* Xtb       = (bf16_t*)alloc((size_t)S_TXT * DIMC * 2);
  bf16_t* WqkvTb    = (bf16_t*)alloc((size_t)DIMC * TDIM * 2);   // [TDIM][DIMC]
  bf16_t* WaddTb    = (bf16_t*)alloc((size_t)DIMC * TDIM * 2);
  bf16_t* WoutTb    = (bf16_t*)alloc((size_t)DIMC * DIMC * 2);
  bf16_t* WaddoutTb = (bf16_t*)alloc((size_t)DIMC * DIMC * 2);
  bf16_t* attnb     = (bf16_t*)alloc((size_t)S_TOT * DIMC * 2);
  bf16_t* qbuf      = (bf16_t*)alloc((size_t)HEADS * S_TOT * HD * 2);
  bf16_t* kbuf      = (bf16_t*)alloc((size_t)HEADS * S_TOT * HD * 2);
  bf16_t* vbuf      = (bf16_t*)alloc((size_t)HEADS * S_TOT * HD * 2);
  float*  img_qkv   = (float*)alloc((size_t)S_IMG * TDIM * 4);
  float*  txt_qkv   = (float*)alloc((size_t)S_TXT * TDIM * 4);
  float*  attn      = (float*)alloc((size_t)S_TOT * DIMC * 4);

  auto cvt = [&](const float* src, bf16_t* dst, int n) {
    const int blocks = (n + 256 * 16 - 1) / (256 * 16);
    k_f32_to_bf16<<<blocks, 256, 0, stream>>>(src, dst, n);
  };
  cvt(image, Xib, S_IMG * DIMC);
  cvt(text,  Xtb, S_TXT * DIMC);
  k_f32_to_bf16_T<<<dim3(TDIM / 32, DIMC / 32), 256, 0, stream>>>(
      Wqkv, WqkvTb, DIMC, TDIM);
  k_f32_to_bf16_T<<<dim3(TDIM / 32, DIMC / 32), 256, 0, stream>>>(
      Wadd, WaddTb, DIMC, TDIM);
  k_f32_to_bf16_T<<<dim3(DIMC / 32, DIMC / 32), 256, 0, stream>>>(
      Wout, WoutTb, DIMC, DIMC);
  k_f32_to_bf16_T<<<dim3(DIMC / 32, DIMC / 32), 256, 0, stream>>>(
      Waddout, WaddoutTb, DIMC, DIMC);

  k_gemm_bf16<<<dim3(TDIM / 128, S_IMG / 128), 256, 0, stream>>>(
      Xib, WqkvTb, bqkv, img_qkv, S_IMG, TDIM, DIMC);
  k_gemm_bf16<<<dim3(TDIM / 128, S_TXT / 128), 256, 0, stream>>>(
      Xtb, WaddTb, badd, txt_qkv, S_TXT, TDIM, DIMC);

  k_prep_qkv<<<dim3(S_TOT, HEADS), 128, 0, stream>>>(
      img_qkv, txt_qkv, img_cos, img_sin, txt_cos, txt_sin,
      gq, gk, gaq, gak, qbuf, kbuf, vbuf);

  k_attention<<<dim3(S_TOT / 64, HEADS), 128, 0, stream>>>(
      qbuf, kbuf, vbuf, attn);

  cvt(attn, attnb, S_TOT * DIMC);

  // img_out = attn[256:, :] @ Wout + bout  -> d_out[0 : 2048*3072)
  k_gemm_bf16<<<dim3(DIMC / 128, S_IMG / 128), 256, 0, stream>>>(
      attnb + (size_t)S_TXT * DIMC, WoutTb, bout, out, S_IMG, DIMC, DIMC);
  // txt_out = attn[:256, :] @ Waddout + baddout -> d_out[2048*3072 : )
  k_gemm_bf16<<<dim3(DIMC / 128, S_TXT / 128), 256, 0, stream>>>(
      attnb, WaddoutTb, baddout, out + (size_t)S_IMG * DIMC, S_TXT, DIMC, DIMC);
}
